// NGramEncoder_53429393162962
// MI455X (gfx1250) — matrix operand order repaired and checked by reading, not compiled, not run on hardware
//
#include <hip/hip_runtime.h>

// Reference dims: S=512, B=64, E=256, H=256 (fp32 in/out)
#define S_TOK   512
#define BATCH   64
#define EDIM    256
#define HDIM    256
#define NSTEPS  1023          // 2*(S-1)+1 sequential cell steps per direction
#define NBLK    16            // 2 dirs x 8 hidden-unit slices
#define A_STR   520           // Ash row stride (bf16 elems): 512 + 8 pad -> 4-bank skew
#define Z_STR   132
#define C_STR   33

typedef __attribute__((ext_vector_type(16))) __bf16   v16bf;
typedef __attribute__((ext_vector_type(8)))  float    v8f;
typedef __attribute__((ext_vector_type(4)))  unsigned uint4v;

union Frag {
  v16bf          v;
  uint4v         q[2];
  unsigned short u[16];
};

__device__ __forceinline__ unsigned short f2bf(float x) {
  unsigned u = __float_as_uint(x);
  u += 0x7fffu + ((u >> 16) & 1u);            // round-to-nearest-even
  return (unsigned short)(u >> 16);
}
__device__ __forceinline__ unsigned pk2bf(float lo, float hi) {
  return (unsigned)f2bf(lo) | ((unsigned)f2bf(hi) << 16);
}
__device__ __forceinline__ float sigm(float x) { return 1.0f / (1.0f + __expf(-x)); }

// ---- CDNA5 async global->LDS DMA (ASYNCcnt), per CDNA5 ISA ----
// GV mode: VDST = LDS byte offset, VADDR = 64-bit global address.
__device__ __forceinline__ void async_g2l_b128(unsigned lds_byte, const void* gaddr) {
  unsigned long long ga = (unsigned long long)gaddr;
  asm volatile("global_load_async_to_lds_b128 %0, %1, off"
               :: "v"(lds_byte), "v"(ga) : "memory");
}
__device__ __forceinline__ void async_wait0() {
  asm volatile("s_wait_asynccnt 0" ::: "memory");
}

// Stage a 64-row x 512-byte contiguous global block into Ash (row stride 1040B)
// at byte column offset `colByte`. 2048 x b128 chunks spread over 128 threads.
__device__ __forceinline__ void stage_async(unsigned ashBase, unsigned colByte,
                                            const unsigned short* gsrc, int tid) {
  #pragma unroll
  for (int i = 0; i < 16; ++i) {
    const int idx = tid + 128 * i;            // 0..2047
    const int row = idx >> 5;                 // 0..63
    const int c   = idx & 31;                 // 0..31 (16B chunks, 512B per row)
    async_g2l_b128(ashBase + (unsigned)(row * (A_STR * 2) + colByte + c * 16),
                   (const char*)gsrc + row * 512 + c * 16);
  }
}

// ---- prep: convert x (S,B,E) f32 -> bf16 once per launch (DMA-able) ----
__global__ void ngram_prep(const float* __restrict__ x, unsigned* __restrict__ xbf) {
  const int n = S_TOK * BATCH * EDIM / 2;
  for (int k = blockIdx.x * blockDim.x + threadIdx.x; k < n;
       k += blockDim.x * gridDim.x) {
    float2 v = ((const float2*)x)[k];
    xbf[k] = pk2bf(v.x, v.y);
  }
}

// ---- init: zero bf16 h double-buffers + barrier counter ----
__global__ void ngram_init(unsigned* hbuf, unsigned* cnt) {
  int i = blockIdx.x * blockDim.x + threadIdx.x;
  const int n = 2 * 2 * BATCH * HDIM / 2;     // uints
  for (int k = i; k < n; k += blockDim.x * gridDim.x) hbuf[k] = 0u;
  if (i == 0) *cnt = 0u;
}

__global__ __launch_bounds__(128, 1)
void ngram_lstm(const unsigned short* __restrict__ xbf,   // (S,B,E) bf16
                const float* __restrict__ wih_f, const float* __restrict__ whh_f,
                const float* __restrict__ bih_f, const float* __restrict__ bhh_f,
                const float* __restrict__ wih_b, const float* __restrict__ whh_b,
                const float* __restrict__ bih_b, const float* __restrict__ bhh_b,
                float* __restrict__ out,                  // (B, 2H) f32
                unsigned short* __restrict__ hbuf,        // [2 buf][2 dir][B][H] bf16
                unsigned* __restrict__ cnt)
{
  __shared__ __align__(16) unsigned short Ash[BATCH * A_STR]; // [x_t | h] bf16, row=batch
  __shared__ float Zsh[BATCH * Z_STR];                        // z slice (64 x 128) f32
  __shared__ float Csh[BATCH * C_STR];                        // local cell state (f32)
  __shared__ float Bias[128];

  const int tid  = (int)threadIdx.x;
  const int lane = tid & 31;
  const int wv   = tid >> 5;                  // wave 0..3
  const int dir  = (int)blockIdx.x >> 3;      // 0 fwd, 1 bwd
  const int u0   = ((int)blockIdx.x & 7) * 32;// hidden-unit slice base
  const int l    = lane & 15;
  const int hi   = lane >> 4;                 // K-half select per WMMA layout

  // low 32 bits of a generic LDS pointer == LDS byte offset (ISA 10.2)
  const unsigned ashBase = (unsigned)(unsigned long long)(void*)Ash;

  const float* wih = dir ? wih_b : wih_f;
  const float* whh = dir ? whh_b : whh_f;
  const float* bih = dir ? bih_b : bih_f;
  const float* bhh = dir ? bhh_b : bhh_f;

  // ---- persist combined [Wih | Whh] slice as WMMA B-fragments in VGPRs ----
  // z-cols n in [0,128): gate = n>>5, unit j = u0 + (n&31) -> weight row gate*H + j
  Frag Bf[32];                                 // [ntLocal(2)][kk(16)]
  #pragma unroll
  for (int i = 0; i < 2; ++i) {
    const int nbase = (wv * 2 + i) * 16;
    const int n     = nbase + l;
    const int row   = (n >> 5) * HDIM + u0 + (n & 31);
    #pragma unroll
    for (int kk = 0; kk < 16; ++kk) {
      const int ks = kk * 32 + hi * 16;        // 16 contiguous K elems (B layout)
      const float* src = (ks < EDIM) ? (wih + (size_t)row * EDIM + ks)
                                     : (whh + (size_t)row * HDIM + (ks - EDIM));
      Frag f;
      #pragma unroll
      for (int q4 = 0; q4 < 4; ++q4) {
        float4 v = *(const float4*)(src + q4 * 4);
        f.u[q4 * 4 + 0] = f2bf(v.x);
        f.u[q4 * 4 + 1] = f2bf(v.y);
        f.u[q4 * 4 + 2] = f2bf(v.z);
        f.u[q4 * 4 + 3] = f2bf(v.w);
      }
      Bf[i * 16 + kk] = f;
    }
  }

  // bias + local cell state init
  {
    const int row = ((tid & 127) >> 5) * 0;    // (silence unused warnings pattern)
    (void)row;
  }
  {
    const int n = tid;                          // 128 threads -> one each
    const int rw = (n >> 5) * HDIM + u0 + (n & 31);
    Bias[n] = bih[rw] + bhh[rw];
  }
  for (int k = tid; k < BATCH * C_STR; k += 128) Csh[k] = 0.0f;

  const int jj   = tid & 31;                   // unit within slice (gate pass)
  const int brow = tid >> 5;                   // 0..3
  float hmax[16];
  #pragma unroll
  for (int r = 0; r < 16; ++r) hmax[r] = -3.0e38f;

  v8f vzero;
  #pragma unroll
  for (int e = 0; e < 8; ++e) vzero[e] = 0.0f;

  // ---- initial stage: x(token of step 0) and h(0) via async DMA ----
  int tprev = (dir == 0) ? 0 : 1;
  stage_async(ashBase, 0, xbf + (size_t)tprev * (BATCH * EDIM), tid);
  stage_async(ashBase, 512, hbuf + (size_t)dir * (BATCH * HDIM), tid); // buf0
  async_wait0();
  __syncthreads();

  for (int s = 0; s < NSTEPS; ++s) {
    // ---- z = [x_t | h] @ Wslice^T : 64x128, K=512, bf16 WMMA, f32 accum ----
    v8f acc[8];
    #pragma unroll
    for (int i = 0; i < 8; ++i) acc[i] = vzero;

    #pragma unroll
    for (int kk = 0; kk < 16; ++kk) {
      Frag a[4];
      #pragma unroll
      for (int mt = 0; mt < 4; ++mt) {
        const unsigned short* ap = &Ash[(mt * 16 + l) * A_STR + kk * 32 + hi * 16];
        a[mt].q[0] = *(const uint4v*)(ap);        // VGPR0-3 K-half
        a[mt].q[1] = *(const uint4v*)(ap + 8);    // VGPR4-7 K-half
      }
      #pragma unroll
      for (int i = 0; i < 2; ++i) {
        #pragma unroll
        for (int mt = 0; mt < 4; ++mt) {
          acc[i * 4 + mt] = __builtin_amdgcn_wmma_f32_16x16x32_bf16(
              false, a[mt].v, false, Bf[i * 16 + kk].v,
              (short)0, acc[i * 4 + mt], false, false);
        }
      }
    }

    // scatter accumulators to z LDS (C/D layout: VGPR r -> row +r+8*hi, col = lane&15)
    #pragma unroll
    for (int i = 0; i < 2; ++i) {
      const int nb = (wv * 2 + i) * 16;
      #pragma unroll
      for (int mt = 0; mt < 4; ++mt) {
        #pragma unroll
        for (int r = 0; r < 8; ++r)
          Zsh[(mt * 16 + hi * 8 + r) * Z_STR + nb + l] = acc[i * 4 + mt][r];
      }
    }
    __syncthreads();   // also: all waves done reading Ash -> safe to restage x below

    // ---- gate pass: i,f,g,o -> c,h ; store h slice (bf16), track running max ----
    unsigned short* hd = hbuf + ((size_t)(((s + 1) & 1) * 2 + dir)) * (BATCH * HDIM);
    #pragma unroll 4
    for (int r = 0; r < 16; ++r) {
      const int b = brow * 16 + r;
      const float zi = Zsh[b * Z_STR +       jj] + Bias[jj];
      const float zf = Zsh[b * Z_STR +  32 + jj] + Bias[32 + jj];
      const float zg = Zsh[b * Z_STR +  64 + jj] + Bias[64 + jj];
      const float zo = Zsh[b * Z_STR +  96 + jj] + Bias[96 + jj];
      float c = Csh[b * C_STR + jj];
      c = sigm(zf) * c + sigm(zi) * tanhf(zg);
      const float h = sigm(zo) * tanhf(c);
      Csh[b * C_STR + jj] = c;
      hd[b * HDIM + u0 + jj] = f2bf(h);
      hmax[r] = fmaxf(hmax[r], h);
    }

    // ---- overlap: issue async restage of x(t_{s+1}) before the global barrier ----
    const int s1 = s + 1;
    if (s1 < NSTEPS) {
      int t;
      if (dir == 0) t = (s1 + (s1 & 1)) >> 1;
      else          t = (s1 == NSTEPS - 1) ? (S_TOK - 1)
                                           : ((s1 & 1) ? (s1 >> 1) : (s1 >> 1) + 1);
      if (t != tprev) {
        stage_async(ashBase, 0, xbf + (size_t)t * (BATCH * EDIM), tid);
        tprev = t;
      }
    }

    // ---- one agent-scope barrier per step (double-buffered h) ----
    __threadfence();
    __syncthreads();
    if (tid == 0) {
      __hip_atomic_fetch_add(cnt, 1u, __ATOMIC_RELEASE, __HIP_MEMORY_SCOPE_AGENT);
      const unsigned want = (unsigned)NBLK * (unsigned)(s + 1);
      while (__hip_atomic_load(cnt, __ATOMIC_ACQUIRE, __HIP_MEMORY_SCOPE_AGENT) < want)
        __builtin_amdgcn_s_sleep(1);
    }
    __syncthreads();
    __threadfence();

    // ---- stage h(s+1) (all slices now published), wait for both DMAs ----
    if (s1 < NSTEPS) {
      stage_async(ashBase, 512,
                  hbuf + ((size_t)((s1 & 1) * 2 + dir)) * (BATCH * HDIM), tid);
      async_wait0();
      __syncthreads();
    }
  }

  // emb = running max over all emitted hidden states
  #pragma unroll
  for (int r = 0; r < 16; ++r) {
    const int b = brow * 16 + r;
    out[b * (2 * HDIM) + dir * HDIM + u0 + jj] = hmax[r];
  }
}

extern "C" void kernel_launch(void* const* d_in, const int* in_sizes, int n_in,
                              void* d_out, int out_size, void* d_ws, size_t ws_size,
                              hipStream_t stream) {
  (void)in_sizes; (void)n_in; (void)out_size; (void)ws_size;
  const float* x     = (const float*)d_in[0];
  const float* wih_f = (const float*)d_in[1];
  const float* whh_f = (const float*)d_in[2];
  const float* bih_f = (const float*)d_in[3];
  const float* bhh_f = (const float*)d_in[4];
  const float* wih_b = (const float*)d_in[5];
  const float* whh_b = (const float*)d_in[6];
  const float* bih_b = (const float*)d_in[7];
  const float* bhh_b = (const float*)d_in[8];

  // workspace layout
  unsigned short* xbf  = (unsigned short*)d_ws;                       // 16 MB
  const size_t    xbsz = (size_t)S_TOK * BATCH * EDIM * sizeof(unsigned short);
  unsigned short* hbuf = (unsigned short*)((char*)d_ws + xbsz);       // 128 KB
  const size_t    hbsz = (size_t)2 * 2 * BATCH * HDIM * sizeof(unsigned short);
  unsigned*       cnt  = (unsigned*)((char*)d_ws + xbsz + hbsz);

  ngram_prep<<<2048, 256, 0, stream>>>(x, (unsigned*)xbf);
  ngram_init<<<32, 256, 0, stream>>>((unsigned*)hbuf, cnt);
  ngram_lstm<<<NBLK, 128, 0, stream>>>(xbf, wih_f, whh_f, bih_f, bhh_f,
                                       wih_b, whh_b, bih_b, bhh_b,
                                       (float*)d_out, hbuf, cnt);
}
